// AbstractAttention_36155034698077
// MI455X (gfx1250) — matrix operand order repaired and checked by reading, not compiled, not run on hardware
//
#include <hip/hip_runtime.h>

// ---------------------------------------------------------------------------
// Multi-head attention forward for MI455X (gfx1250, wave32, WMMA).
// B=2, S=2048, D_MODEL=1024, H=16, d_k=64 (S/B derived at launch).
// Precision: bf16 operands, fp32 accumulation (v_wmma_f32_16x16x32_bf16).
// Data movement: TDM tensor_load_to_lds for K tiles, async global->LDS for
// byte-copy stages, VGPR staging where f32->bf16 conversion or the V interleave is needed.
// ---------------------------------------------------------------------------

#define DMODEL 1024
#define NHEADS 16
#define DK     64

typedef __attribute__((ext_vector_type(16))) __bf16 v16bf;
typedef __attribute__((ext_vector_type(8)))  float  v8f;
typedef __attribute__((ext_vector_type(4)))  unsigned int u32x4;
typedef __attribute__((ext_vector_type(8)))  int          i32x8;
typedef __attribute__((ext_vector_type(4)))  int          i32x4;

// GCC-style vector types for the async-LDS builtin pointer params (clang
// prints the parameter as 'int __attribute__((vector_size(16))) *').
typedef int v4i_b __attribute__((vector_size(16)));
typedef __attribute__((address_space(3))) v4i_b* lds_v4i_p;

union V16U {
    v16bf v;
    unsigned int d[8];
    unsigned short u[16];
};

__device__ __forceinline__ unsigned short f2bf(float f) {
    unsigned int u = __float_as_uint(f);
    u += 0x7fffu + ((u >> 16) & 1u);      // round-to-nearest-even
    return (unsigned short)(u >> 16);
}
__device__ __forceinline__ unsigned int pack2bf(float lo, float hi) {
    return (unsigned int)f2bf(lo) | ((unsigned int)f2bf(hi) << 16);
}
__device__ __forceinline__ v8f wmma_bf16(const V16U& a, const V16U& b, v8f c) {
    return __builtin_amdgcn_wmma_f32_16x16x32_bf16(
        false, a.v, false, b.v, (short)0, c, false, false);
}

// ---- CDNA5 async global->LDS copy (ASYNCcnt) with portable fallback -------
#if __has_builtin(__builtin_amdgcn_global_load_async_to_lds_b128) && \
    __has_builtin(__builtin_amdgcn_s_wait_asynccnt)
#define USE_ASYNC_LDS 1
#else
#define USE_ASYNC_LDS 0
#endif

__device__ __forceinline__ void cp_b128_to_lds(void* lds, const void* g) {
#if USE_ASYNC_LDS
    __builtin_amdgcn_global_load_async_to_lds_b128(
        (v4i_b*)const_cast<void*>(g), (lds_v4i_p)lds, 0, 0);
#else
    *(uint4*)lds = *(const uint4*)g;
#endif
}
__device__ __forceinline__ void cp_async_wait() {
#if USE_ASYNC_LDS
    __builtin_amdgcn_s_wait_asynccnt(0);
#endif
}

// ---- CDNA5 Tensor Data Mover (TENSORcnt) with portable fallback -----------
#if __has_builtin(__builtin_amdgcn_tensor_load_to_lds) && \
    __has_builtin(__builtin_amdgcn_s_wait_tensorcnt)
#define USE_TDM 1
#else
#define USE_TDM 0
#endif

__device__ __forceinline__ unsigned lds_byte_off(const void* p) {
    return (unsigned)(unsigned long long)
        (__attribute__((address_space(3))) const void*)p;
}

#if USE_TDM
// 2D bf16 tile load: tile_dim0 x tile_dim1 elements, tensor_dim0_stride = row
// pitch (elements), LDS padding: pad_amount DWORDs every pad_interval DWORDs.
// pad_interval_code: interval = 2^(code+1) DWORDs. pad_amount_code: amount =
// code+1 DWORDs (0 disables padding when both are 0).
__device__ __forceinline__ void tdm_load_tile_2d_bf16(
    const void* gsrc, unsigned lds_addr, int tensor_d0, int tensor_d1,
    int tile_d0, int tile_d1, int d0_stride,
    unsigned pad_interval_code, unsigned pad_amount_code)
{
    unsigned long long ga = (unsigned long long)gsrc;
    u32x4 g0;
    g0[0] = 1u;                                        // count=1, user mode
    g0[1] = lds_addr;                                  // lds_addr [63:32]
    g0[2] = (unsigned)ga;                              // global_addr lo
    g0[3] = (unsigned)((ga >> 32) & 0x1ffffffu) | (2u << 30);  // addr hi|type=2
    const unsigned pad_en = (pad_amount_code | pad_interval_code) ? 1u : 0u;
    i32x8 g1;
    g1[0] = (int)((1u << 16)                           // data_size = 2 bytes
                | (pad_en << 20)
                | (pad_interval_code << 22)
                | (pad_amount_code << 25));
    g1[1] = (tensor_d0 & 0xffff) << 16;                // tensor_dim0 lo16
    g1[2] = ((tensor_d0 >> 16) & 0xffff)               // tensor_dim0 hi16
          | ((tensor_d1 & 0xffff) << 16);              // tensor_dim1 lo16
    g1[3] = ((tensor_d1 >> 16) & 0xffff)               // tensor_dim1 hi16
          | ((tile_d0 & 0xffff) << 16);                // tile_dim0
    g1[4] = (tile_d1 & 0xffff);                        // tile_dim1 (tile_dim2=0)
    g1[5] = d0_stride;                                 // tensor_dim0_stride lo32
    g1[6] = 0;
    g1[7] = 0;
    const i32x4 z4 = {0, 0, 0, 0};
#if defined(__clang_major__) && __clang_major__ >= 23
    const i32x8 z8 = {0, 0, 0, 0, 0, 0, 0, 0};
    __builtin_amdgcn_tensor_load_to_lds(g0, g1, z4, z4, z8, 0);
#else
    __builtin_amdgcn_tensor_load_to_lds(g0, g1, z4, z4, 0);
#endif
}
#endif // USE_TDM

// ---------------------------------------------------------------------------
// GEMM: C[m][n] = sum_k A[m][k] * W[n][k] + bias[n]    (M x 1024 x 1024)
// A_BF16: A is bf16 (merged attention output, async-copied to LDS);
//         else fp32 (converted to bf16 through VGPRs).
// OUT_HEADSPLIT: store bf16 into [B,H,S,64]; else fp32 row-major to d_out.
// Block tile 128x128x32, 256 threads = 8 waves (2 along M x 4 along N),
// each wave computes 4x2 = 8 WMMA tiles of 16x16.
// ---------------------------------------------------------------------------
template <bool A_BF16, bool OUT_HEADSPLIT>
__global__ __launch_bounds__(256) void gemm_wmma_kernel(
    const float* __restrict__ Af, const unsigned short* __restrict__ Ab,
    const float* __restrict__ W, const float* __restrict__ bias,
    unsigned short* __restrict__ outB, float* __restrict__ outF,
    int M, int S)
{
    const int tid  = threadIdx.x;
    const int wave = tid >> 5;
    const int lane = tid & 31;
    const int half = lane >> 4;
    const int l16  = lane & 15;
    const int wm   = wave & 1;   // 2 waves along M (64 rows each)
    const int wn   = wave >> 1;  // 4 waves along N (32 cols each)
    const int m0   = blockIdx.x * 128;
    const int n0   = blockIdx.y * 128;

    // A: [128][32] bf16, padded row stride 40 (80 B -> 16 B-aligned chunks)
    __shared__ unsigned short ldsA[128 * 40];
    // B: interleaved [k2=16][n=128(+4 pad)][2] so the (k,k+1) pair is one b32
    __shared__ unsigned short ldsB[16 * 132 * 2];

    const v8f vzero = {0.f, 0.f, 0.f, 0.f, 0.f, 0.f, 0.f, 0.f};
    v8f acc[4][2];
#pragma unroll
    for (int mt = 0; mt < 4; ++mt)
#pragma unroll
        for (int nt = 0; nt < 2; ++nt) acc[mt][nt] = vzero;

    for (int k0 = 0; k0 < DMODEL; k0 += 32) {
        __syncthreads();
        // ---- stage A tile: 128 rows x 32 k
        if (A_BF16) {
            // pure byte copy -> async global->LDS b128 (2 chunks per thread)
#pragma unroll
            for (int i = 0; i < 2; ++i) {
                int lin = tid + 256 * i;            // 0..511
                int row = lin >> 2, c8 = lin & 3;   // 4 x 16B per row
                cp_b128_to_lds(ldsA + row * 40 + c8 * 8,
                               Ab + (size_t)(m0 + row) * DMODEL + k0 + c8 * 8);
            }
        } else {
#pragma unroll
            for (int i = 0; i < 8; ++i) {
                int lin = tid + 256 * i;            // 0..2047
                int row = lin >> 4, c2 = lin & 15;  // k = 2*c2
                const float2 f2 =
                    *(const float2*)(Af + (size_t)(m0 + row) * DMODEL + k0 + c2 * 2);
                *(unsigned int*)(ldsA + row * 40 + c2 * 2) = pack2bf(f2.x, f2.y);
            }
        }
        // ---- stage B tile (weights): W[n0+n][k0+2*c2 .. +1] -> interleaved
#pragma unroll
        for (int i = 0; i < 8; ++i) {
            int lin = tid + 256 * i;
            int n = lin >> 4, c2 = lin & 15;
            const float2 f2 =
                *(const float2*)(W + (size_t)(n0 + n) * DMODEL + k0 + c2 * 2);
            *(unsigned int*)(ldsB + (c2 * 132 + n) * 2) = pack2bf(f2.x, f2.y);
        }
        // ---- speculative prefetch of the next K-slab (global_prefetch_b8)
        if (k0 + 32 < DMODEL) {
            int row = tid >> 4, c2 = tid & 15;
            if (A_BF16)
                __builtin_prefetch(Ab + (size_t)(m0 + row) * DMODEL + (k0 + 32) + c2 * 2, 0, 3);
            else
                __builtin_prefetch(Af + (size_t)(m0 + row) * DMODEL + (k0 + 32) + c2 * 2, 0, 3);
            __builtin_prefetch(W + (size_t)(n0 + row) * DMODEL + (k0 + 32) + c2 * 2, 0, 3);
        }
        if (A_BF16) cp_async_wait();
        __syncthreads();

        // ---- B fragments (32x16): VGPR v holds K = half*16 + 2v, 2v+1; N = l16
        V16U bfr[2];
#pragma unroll
        for (int nt = 0; nt < 2; ++nt) {
            int n = wn * 32 + nt * 16 + l16;
#pragma unroll
            for (int v = 0; v < 8; ++v) {
                int k2 = half * 8 + v;
                bfr[nt].d[v] = *(const unsigned int*)(ldsB + (k2 * 132 + n) * 2);
            }
        }
        // ---- A fragments (16x32) + WMMA
#pragma unroll
        for (int mt = 0; mt < 4; ++mt) {
            V16U afr;
            int row = wm * 64 + mt * 16 + l16;
#pragma unroll
            for (int v = 0; v < 8; ++v) {
                int kk = half * 16 + 2 * v;
                afr.d[v] = *(const unsigned int*)(ldsA + row * 40 + kk);
            }
#pragma unroll
            for (int nt = 0; nt < 2; ++nt)
                acc[mt][nt] = wmma_bf16(afr, bfr[nt], acc[mt][nt]);
        }
    }

    // ---- epilogue: bias, then store (C layout: row = half*8+v, col = l16)
#pragma unroll
    for (int mt = 0; mt < 4; ++mt)
#pragma unroll
        for (int nt = 0; nt < 2; ++nt)
#pragma unroll
            for (int v = 0; v < 8; ++v) {
                int row = m0 + wm * 64 + mt * 16 + half * 8 + v;  // global m
                int col = n0 + wn * 32 + nt * 16 + l16;           // global n
                float val = acc[mt][nt][v] + bias[col];
                if (OUT_HEADSPLIT) {
                    int b = row / S, s = row % S;
                    int h = col >> 6, d = col & 63;
                    outB[(((size_t)(b * NHEADS + h)) * S + s) * DK + d] = f2bf(val);
                } else {
                    outF[(size_t)row * DMODEL + col] = val;
                }
            }
}

// ---------------------------------------------------------------------------
// Fused flash attention: softmax(Q Kt / sqrt(dk) + mask) V per (b, h).
// One workgroup = 128 query rows (8 waves x 16 rows); KV streamed 64 keys
// at a time through LDS. Q fragments live in registers in exact A-layout.
// K tile staged by the Tensor Data Mover (wave 0, TENSORcnt); mask tile by
// async global->LDS (ASYNCcnt); V tile through VGPRs (layout interleave).
// ---------------------------------------------------------------------------
__global__ __launch_bounds__(256) void attn_wmma_kernel(
    const unsigned short* __restrict__ Qh,
    const unsigned short* __restrict__ Kh,
    const unsigned short* __restrict__ Vh,
    const unsigned char* __restrict__ mask,
    unsigned short* __restrict__ X2,
    int S)
{
    const int tid  = threadIdx.x;
    const int wave = tid >> 5;
    const int lane = tid & 31;
    const int half = lane >> 4;
    const int l16  = lane & 15;

    const int nq   = S >> 7;               // 128-row q blocks
    const int qblk = blockIdx.x % nq;
    const int bh   = blockIdx.x / nq;
    const int b    = bh / NHEADS;

    __shared__ unsigned short ldsK[64 * 72];        // [key][d], stride 72
    __shared__ unsigned short ldsV[32 * 132];       // [key>>1][d][key&1] interleaved
    __shared__ unsigned short ldsP[8][16 * 72];     // per-wave P scratch [row][key]
    __shared__ unsigned char  ldsM[128 * 64];       // mask tile [qrow][key]

    const size_t headBase = (size_t)bh * S * DK;

    // ---- load Q fragments (16 rows x 64 d = two 16x32 A fragments)
    V16U qf[2];
    {
        int qrow = qblk * 128 + wave * 16 + l16;
        const unsigned short* qp = Qh + headBase + (size_t)qrow * DK;
#pragma unroll
        for (int f = 0; f < 2; ++f)
#pragma unroll
            for (int v = 0; v < 8; ++v) {
                int kk = f * 32 + half * 16 + 2 * v;
                qf[f].d[v] = *(const unsigned int*)(qp + kk);
            }
    }

    const v8f vzero = {0.f, 0.f, 0.f, 0.f, 0.f, 0.f, 0.f, 0.f};
    v8f O[4];
#pragma unroll
    for (int dt = 0; dt < 4; ++dt) O[dt] = vzero;
    float mrow[8], lrow[8];
#pragma unroll
    for (int v = 0; v < 8; ++v) { mrow[v] = -1e30f; lrow[v] = 0.f; }

    const int nkb = S >> 6;                 // 64-key blocks
    for (int kb = 0; kb < nkb; ++kb) {
        const int key0 = kb * 64;
        __syncthreads();
        // ---- stage K tile: 64 keys x 64 d bf16 into stride-72 rows
#if USE_TDM
        if (wave == 0) {
            // pad: 4 DWORDs (8 ushorts) after every 32 DWORDs (one 64-elem row)
            // -> LDS row pitch 72 ushorts, matching ldsK.
            tdm_load_tile_2d_bf16(
                Kh + headBase + (size_t)key0 * DK, lds_byte_off(ldsK),
                /*tensor_d0=*/DK, /*tensor_d1=*/S,
                /*tile_d0=*/DK, /*tile_d1=*/64, /*d0_stride=*/DK,
                /*pad_interval_code=*/4u, /*pad_amount_code=*/3u);
        }
#else
#pragma unroll
        for (int i = 0; i < 2; ++i) {
            int lin = tid + 256 * i;        // 0..511
            int row = lin >> 3, c8 = lin & 7;
            cp_b128_to_lds(ldsK + row * 72 + c8 * 8,
                           Kh + headBase + (size_t)(key0 + row) * DK + c8 * 8);
        }
#endif
        // ---- stage V tile, interleaved so (key,key+1) pairs are adjacent
#pragma unroll
        for (int i = 0; i < 4; ++i) {
            int lin = tid + 256 * i;        // 0..1023
            int row = lin >> 4, c4 = lin & 15;
            uint2 val = *(const uint2*)(Vh + headBase + (size_t)(key0 + row) * DK + c4 * 4);
            unsigned short* vb = ldsV + (row >> 1) * 132 + (row & 1);
            vb[2 * (c4 * 4 + 0)] = (unsigned short)(val.x & 0xffffu);
            vb[2 * (c4 * 4 + 1)] = (unsigned short)(val.x >> 16);
            vb[2 * (c4 * 4 + 2)] = (unsigned short)(val.y & 0xffffu);
            vb[2 * (c4 * 4 + 3)] = (unsigned short)(val.y >> 16);
        }
        // ---- stage mask tile (bool bytes): 128 q rows x 64 keys (async)
        {
            const unsigned char* mb =
                mask + (size_t)b * S * S + (size_t)(qblk * 128) * S + key0;
#pragma unroll
            for (int i = 0; i < 2; ++i) {
                int lin = tid + 256 * i;    // 0..511
                int row = lin >> 2, c16 = lin & 3;
                cp_b128_to_lds(ldsM + row * 64 + c16 * 16,
                               mb + (size_t)row * S + c16 * 16);
            }
        }
        cp_async_wait();                    // our async writes visible
#if USE_TDM
        if (wave == 0) __builtin_amdgcn_s_wait_tensorcnt(0);
#endif
        __syncthreads();

        // ---- scores: 16 x 64, four 16x16 fragments, 2 WMMAs each (d_k=64)
        v8f sf[4];
#pragma unroll
        for (int nt = 0; nt < 4; ++nt) {
            v8f z = vzero;
            int key = nt * 16 + l16;
#pragma unroll
            for (int f = 0; f < 2; ++f) {
                V16U kfr;
#pragma unroll
                for (int v = 0; v < 8; ++v) {
                    int dd = f * 32 + half * 16 + 2 * v;
                    kfr.d[v] = *(const unsigned int*)(ldsK + key * 72 + dd);
                }
                z = wmma_bf16(qf[f], kfr, z);
            }
            sf[nt] = z;
        }

        // ---- scale (1/sqrt(64)=0.125) + mask
#pragma unroll
        for (int nt = 0; nt < 4; ++nt)
#pragma unroll
            for (int v = 0; v < 8; ++v) {
                int qr = wave * 16 + half * 8 + v;
                int kc = nt * 16 + l16;
                float s = sf[nt][v] * 0.125f;
                if (!ldsM[qr * 64 + kc]) s = -1e9f;
                sf[nt][v] = s;
            }

        // ---- online softmax (rows live in 16-lane halves of the wave)
        float corr[8];
#pragma unroll
        for (int v = 0; v < 8; ++v) {
            float x = sf[0][v];
#pragma unroll
            for (int nt = 1; nt < 4; ++nt) x = fmaxf(x, sf[nt][v]);
            x = fmaxf(x, __shfl_xor(x, 1, 32));
            x = fmaxf(x, __shfl_xor(x, 2, 32));
            x = fmaxf(x, __shfl_xor(x, 4, 32));
            x = fmaxf(x, __shfl_xor(x, 8, 32));
            float nm = fmaxf(mrow[v], x);
            corr[v] = __expf(mrow[v] - nm);
            float rs = 0.f;
#pragma unroll
            for (int nt = 0; nt < 4; ++nt) {
                float p = __expf(sf[nt][v] - nm);
                sf[nt][v] = p;
                rs += p;
            }
            rs += __shfl_xor(rs, 1, 32);
            rs += __shfl_xor(rs, 2, 32);
            rs += __shfl_xor(rs, 4, 32);
            rs += __shfl_xor(rs, 8, 32);
            lrow[v] = lrow[v] * corr[v] + rs;
            mrow[v] = nm;
        }
#pragma unroll
        for (int dt = 0; dt < 4; ++dt)
#pragma unroll
            for (int v = 0; v < 8; ++v) O[dt][v] *= corr[v];

        // ---- P: C-layout -> A-layout via wave-private LDS scratch
        unsigned short* pw = ldsP[wave];
#pragma unroll
        for (int nt = 0; nt < 4; ++nt)
#pragma unroll
            for (int v = 0; v < 8; ++v)
                pw[(half * 8 + v) * 72 + nt * 16 + l16] = f2bf(sf[nt][v]);

        // ---- O += P @ V  (two 16x32 P fragments over the 64 keys)
#pragma unroll
        for (int kf = 0; kf < 2; ++kf) {
            V16U pfr;
#pragma unroll
            for (int v = 0; v < 8; ++v) {
                int kk = kf * 32 + half * 16 + 2 * v;
                pfr.d[v] = *(const unsigned int*)(pw + l16 * 72 + kk);
            }
#pragma unroll
            for (int dt = 0; dt < 4; ++dt) {
                V16U vfr;
#pragma unroll
                for (int v = 0; v < 8; ++v) {
                    int key = kf * 32 + half * 16 + 2 * v;      // even
                    int dd = dt * 16 + l16;
                    vfr.d[v] = *(const unsigned int*)(ldsV + (key >> 1) * 132 + 2 * dd);
                }
                O[dt] = wmma_bf16(pfr, vfr, O[dt]);
            }
        }
    }

    // ---- epilogue: normalize, merge heads into X2[b][s][h*64+d] (bf16)
    const int h = bh % NHEADS;
#pragma unroll
    for (int dt = 0; dt < 4; ++dt)
#pragma unroll
        for (int v = 0; v < 8; ++v) {
            int q = qblk * 128 + wave * 16 + half * 8 + v;
            int d = dt * 16 + l16;
            float val = O[dt][v] / lrow[v];
            X2[((size_t)b * S + q) * DMODEL + h * DK + d] = f2bf(val);
        }
}

// ---------------------------------------------------------------------------
// Host launcher
// ---------------------------------------------------------------------------
extern "C" void kernel_launch(void* const* d_in, const int* in_sizes, int n_in,
                              void* d_out, int out_size, void* d_ws, size_t ws_size,
                              hipStream_t stream) {
    const float* query = (const float*)d_in[0];
    const float* key   = (const float*)d_in[1];
    const float* value = (const float*)d_in[2];
    const unsigned char* mask = (const unsigned char*)d_in[3];
    const float* Wq = (const float*)d_in[4];
    const float* bq = (const float*)d_in[5];
    const float* Wk = (const float*)d_in[6];
    const float* bk = (const float*)d_in[7];
    const float* Wv = (const float*)d_in[8];
    const float* bv = (const float*)d_in[9];
    const float* Wo = (const float*)d_in[10];
    const float* bo = (const float*)d_in[11];

    const long long qsz = in_sizes[0];              // B*S*D
    const long long msz = in_sizes[3];              // B*S*S
    const int S = (int)((msz * (long long)DMODEL) / qsz);
    const int B = (int)(qsz / ((long long)S * DMODEL));
    const int M = B * S;

    unsigned short* Qh = (unsigned short*)d_ws;     // [B,H,S,64] bf16
    unsigned short* Kh = Qh + (size_t)M * DMODEL;
    unsigned short* Vh = Kh + (size_t)M * DMODEL;
    unsigned short* X2 = Vh + (size_t)M * DMODEL;   // [B,S,D] bf16 merged heads

    dim3 grid(M / 128, DMODEL / 128), block(256);

    gemm_wmma_kernel<false, true><<<grid, block, 0, stream>>>(
        query, nullptr, Wq, bq, Qh, nullptr, M, S);
    gemm_wmma_kernel<false, true><<<grid, block, 0, stream>>>(
        key, nullptr, Wk, bk, Kh, nullptr, M, S);
    gemm_wmma_kernel<false, true><<<grid, block, 0, stream>>>(
        value, nullptr, Wv, bv, Vh, nullptr, M, S);

    const int nAttn = B * NHEADS * (S / 128);
    attn_wmma_kernel<<<nAttn, block, 0, stream>>>(Qh, Kh, Vh, mask, X2, S);

    gemm_wmma_kernel<true, false><<<grid, block, 0, stream>>>(
        nullptr, X2, Wo, bo, nullptr, (float*)d_out, M, S);
}
